// AttentionSelector_32667521254025
// MI455X (gfx1250) — matrix operand order repaired and checked by reading, not compiled, not run on hardware
//
#include <hip/hip_runtime.h>
#include <hip/hip_fp16.h>

typedef __attribute__((ext_vector_type(16))) _Float16 v16h;
typedef __attribute__((ext_vector_type(8)))  _Float16 v8h;
typedef __attribute__((ext_vector_type(8)))  float    v8f;

#define N_PAIRS   65536
#define N_QUERY   8192
#define DIM       66
#define DPK       96         // padded Q/K dim: 3 x K=32 WMMA steps
#define XP        80         // padded X dim: 66 data + ones col @66 + zeros (5 x 16 blocks)
#define KT        64         // keys per mainloop iteration
#define NSPLIT    8
#define KEYS_PER_SPLIT (N_PAIRS / NSPLIT)
#define ITERS     (KEYS_PER_SPLIT / KT)           // 128
#define KL_PAD    104        // K LDS row pitch (halves): 52 banks -> conflict-free
#define XL_PAD    72         // X LDS row pitch (halves): 36 banks -> conflict-free
#define PL_PAD    72
#define PART_STRIDE (16*XP + 2*16)                // 1312 floats: O[16][80], m[16], l[16]

// per-lane async copy: 16B global -> 16B LDS, tracked by ASYNCcnt
#define ASYNC_B128(ldsa, ga) \
    asm volatile("global_load_async_to_lds_b128 %0, %1, off" :: "v"(ldsa), "v"(ga) : "memory")

__device__ __forceinline__ unsigned lds_off(const void* p) {
    // generic LDS pointer: addr[31:0] is the LDS byte offset (aperture in high bits)
    return (unsigned)(unsigned long long)p;
}

__device__ __forceinline__ v16h ld_b32x2(const _Float16* p0, const _Float16* p1) {
    v8h a = *(const v8h*)p0;
    v8h b = *(const v8h*)p1;
    return __builtin_shufflevector(a, b, 0,1,2,3,4,5,6,7,8,9,10,11,12,13,14,15);
}

// ---------------------------------------------------------------------------
// Phase 1: K = x @ Wk^T + bk -> f16 padded to 96 ; query -> f16 padded to 96
// ---------------------------------------------------------------------------
__global__ void prep_kernel(const float* __restrict__ x, const float* __restrict__ query,
                            const float* __restrict__ Wk, const float* __restrict__ bk,
                            _Float16* __restrict__ Kh, _Float16* __restrict__ qh) {
    int gid = blockIdx.x * blockDim.x + threadIdx.x;
    const int totalK = N_PAIRS * DPK;
    if (gid < totalK) {
        int n = gid / DPK, d = gid % DPK;
        float acc = 0.f;
        if (d < DIM) {
            acc = bk[d];
            const float* xr = x + (size_t)n * DIM;
            const float* wr = Wk + (size_t)d * DIM;    // Wk 17KB -> cache resident
            #pragma unroll 6
            for (int j = 0; j < DIM; ++j) acc += xr[j] * wr[j];
        }
        Kh[gid] = (_Float16)acc;
    } else {
        int g = gid - totalK;
        if (g < N_QUERY * DPK) {
            int n = g / DPK, d = g % DPK;
            qh[g] = (_Float16)(d < DIM ? query[(size_t)n * DIM + d] : 0.f);
        }
    }
}

// ---------------------------------------------------------------------------
// Phase 2: xT[d][n] = x[n][d] f16; row 66 = 1.0 (fused row-sum column); 67..79 = 0
// ---------------------------------------------------------------------------
__global__ void xpose_kernel(const float* __restrict__ x, _Float16* __restrict__ xT) {
    __shared__ float tile[32][33];
    int nb = blockIdx.x * 32;
    int db = blockIdx.y * 32;
    int tx = threadIdx.x, ty = threadIdx.y;
    for (int i = ty; i < 32; i += 8) {
        int d = db + tx;
        tile[i][tx] = (d < DIM) ? x[(size_t)(nb + i) * DIM + d] : 0.f;
    }
    __syncthreads();
    for (int i = ty; i < 32; i += 8) {
        int d = db + i;
        if (d < XP) {
            float v = (d < DIM) ? tile[tx][i] : (d == DIM ? 1.f : 0.f);
            xT[(size_t)d * N_PAIRS + (nb + tx)] = (_Float16)v;
        }
    }
}

// ---------------------------------------------------------------------------
// Phase 3: flash attention. 4 waves/WG share async-LDS-staged K/X tiles
// (double buffered); each wave owns a 16-query tile. 22 WMMAs / 64 keys.
// B operands are register-double-buffered so DS latency hides under WMMAs.
// ---------------------------------------------------------------------------
__global__ void __launch_bounds__(128)
flash_kernel(const _Float16* __restrict__ Kh, const _Float16* __restrict__ qh,
             const _Float16* __restrict__ xT, float* __restrict__ part) {
    __shared__ __align__(16) _Float16 Kl[2][KT][KL_PAD];   // 26.6 KB
    __shared__ __align__(16) _Float16 Xl[2][XP][XL_PAD];   // 23.0 KB
    __shared__ __align__(16) _Float16 Pl[4][16][PL_PAD];   //  9.2 KB
    const int tid   = threadIdx.x;
    const int wave  = tid >> 5;
    const int lane  = tid & 31;
    const int l15   = lane & 15;
    const bool hi   = lane >= 16;
    const int split = blockIdx.y;
    const int qt    = blockIdx.x * 4 + wave;     // 16-row query tile id
    const int m0    = qt * 16;

    // --- precomputed per-thread async-copy offsets (bytes) ---
    unsigned kL[6]; unsigned long long kG[6];    // K tile: 768 x16B chunks, 6/thread
    #pragma unroll
    for (int i = 0; i < 6; ++i) {
        int c = tid + 128 * i, r = c / 12, j = c % 12;
        kL[i] = r * (KL_PAD * 2) + j * 16;
        kG[i] = (unsigned long long)r * (DPK * 2) + j * 16;
    }
    unsigned xL[5]; unsigned long long xG[5];    // X tile: 640 x16B chunks, 5/thread
    #pragma unroll
    for (int i = 0; i < 5; ++i) {
        int c = tid + 128 * i, d = c / 8, j = c % 8;
        xL[i] = d * (XL_PAD * 2) + j * 16;
        xG[i] = (unsigned long long)d * ((size_t)N_PAIRS * 2) + j * 16;
    }
    const unsigned kBase[2] = { lds_off(&Kl[0][0][0]), lds_off(&Kl[1][0][0]) };
    const unsigned xBase[2] = { lds_off(&Xl[0][0][0]), lds_off(&Xl[1][0][0]) };

    auto prefetch = [&](int buf, int nb) {
        const char* kg = (const char*)Kh + (size_t)nb * (DPK * 2);
        #pragma unroll
        for (int i = 0; i < 6; ++i) ASYNC_B128(kBase[buf] + kL[i], kg + kG[i]);
        const char* xg = (const char*)xT + (size_t)nb * 2;
        #pragma unroll
        for (int i = 0; i < 5; ++i) ASYNC_B128(xBase[buf] + xL[i], xg + xG[i]);
    };

    // --- Q as A-operand, 3 K-steps (A 16x32: lane<16 K{0..15}, lane>=16 K{16..31})
    v16h aq[3];
    #pragma unroll
    for (int s = 0; s < 3; ++s) {
        const _Float16* qb = qh + (size_t)(m0 + l15) * DPK + s * 32 + (hi ? 16 : 0);
        aq[s] = ld_b32x2(qb, qb + 8);
    }

    v8f O[5];
    #pragma unroll
    for (int b = 0; b < 5; ++b) O[b] = (v8f){0.f,0.f,0.f,0.f,0.f,0.f,0.f,0.f};
    float mrow[8];
    #pragma unroll
    for (int r = 0; r < 8; ++r) mrow[r] = -__builtin_inff();

    const int nb0 = split * KEYS_PER_SPLIT;
    prefetch(0, nb0);

    for (int it = 0; it < ITERS; ++it) {
        const int cur = it & 1;
        if (it + 1 < ITERS) {
            prefetch(cur ^ 1, nb0 + (it + 1) * KT);
            asm volatile("s_wait_asynccnt 0xb" ::: "memory");  // first 11 (cur buffer) done
        } else {
            asm volatile("s_wait_asynccnt 0x0" ::: "memory");
        }
        __syncthreads();                                       // cur tile visible to all waves

        // --- scores: 4 key-subtiles x 3 K-steps; B operands double-buffered in regs
        const int ko = hi ? 16 : 0;
        v8f c[4];
        #pragma unroll
        for (int j = 0; j < 4; ++j) c[j] = (v8f){0.f,0.f,0.f,0.f,0.f,0.f,0.f,0.f};
        v16h bcur[4], bnxt[4];
        #pragma unroll
        for (int j = 0; j < 4; ++j) {
            const _Float16* kb = &Kl[cur][j * 16 + l15][ko];
            bcur[j] = ld_b32x2(kb, kb + 8);
        }
        #pragma unroll
        for (int s = 0; s < 3; ++s) {
            if (s < 2) {
                #pragma unroll
                for (int j = 0; j < 4; ++j) {
                    const _Float16* kb = &Kl[cur][j * 16 + l15][(s + 1) * 32 + ko];
                    bnxt[j] = ld_b32x2(kb, kb + 8);
                }
            }
            #pragma unroll
            for (int j = 0; j < 4; ++j)
                c[j] = __builtin_amdgcn_wmma_f32_16x16x32_f16(false, aq[s], false, bcur[j],
                                                              (short)0, c[j], false, false);
            #pragma unroll
            for (int j = 0; j < 4; ++j) bcur[j] = bnxt[j];
        }

        // --- online softmax stats (row max only; row sum rides in O via ones-column)
        float mo[8];
        int upd = 0;
        #pragma unroll
        for (int r = 0; r < 8; ++r) {
            float t = fmaxf(fmaxf(c[0][r], c[1][r]), fmaxf(c[2][r], c[3][r]));
            t = fmaxf(t, __shfl_xor(t, 1, 32));
            t = fmaxf(t, __shfl_xor(t, 2, 32));
            t = fmaxf(t, __shfl_xor(t, 4, 32));
            t = fmaxf(t, __shfl_xor(t, 8, 32));      // row max within 16-lane half
            mo[r] = mrow[r];
            upd |= (t > mo[r]);
            mrow[r] = fmaxf(mo[r], t);
        }
        if (__any(upd)) {                             // uniform branch; rare after warm-up
            #pragma unroll
            for (int r = 0; r < 8; ++r) {
                float sc = __expf(mo[r] - mrow[r]);
                #pragma unroll
                for (int b = 0; b < 5; ++b) O[b][r] *= sc;
            }
        }

        // --- X B-operands for h=0 issued BEFORE the P staging, so the store-drain
        //     wait also retires them and WMMAs can start immediately after.
        v16h bxA[5], bxB[5];
        #pragma unroll
        for (int b = 0; b < 5; ++b) {
            const _Float16* xb = &Xl[cur][b * 16 + l15][ko];
            bxA[b] = ld_b32x2(xb, xb + 8);
        }

        // --- P = exp(S - m) -> f16, re-layout (C layout -> A layout) through LDS
        #pragma unroll
        for (int r = 0; r < 8; ++r) {
            int row = r + (hi ? 8 : 0);
            #pragma unroll
            for (int j = 0; j < 4; ++j)
                Pl[wave][row][j * 16 + l15] = (_Float16)__expf(c[j][r] - mrow[r]);
        }
        asm volatile("s_wait_dscnt 0x0" ::: "memory");   // P stores drained (per-wave in-order DS)

        // --- load both P A-operands and the h=1 X operands up front
        const _Float16* pb0 = &Pl[wave][l15][hi ? 16 : 0];
        v16h pa0 = ld_b32x2(pb0, pb0 + 8);
        const _Float16* pb1 = pb0 + 32;
        v16h pa1 = ld_b32x2(pb1, pb1 + 8);
        #pragma unroll
        for (int b = 0; b < 5; ++b) {
            const _Float16* xb = &Xl[cur][b * 16 + l15][32 + ko];
            bxB[b] = ld_b32x2(xb, xb + 8);
        }

        // --- O += P @ Xtile : 5 dim-blocks x 2 K=32 halves
        #pragma unroll
        for (int b = 0; b < 5; ++b)
            O[b] = __builtin_amdgcn_wmma_f32_16x16x32_f16(false, pa0, false, bxA[b],
                                                          (short)0, O[b], false, false);
        #pragma unroll
        for (int b = 0; b < 5; ++b)
            O[b] = __builtin_amdgcn_wmma_f32_16x16x32_f16(false, pa1, false, bxB[b],
                                                          (short)0, O[b], false, false);

        __syncthreads();                              // all waves done reading cur tile
    }

    // --- write partial (O, m, l). l = ones-column = block 4, col 2 (dim 66)
    float* pd = part + ((size_t)qt * NSPLIT + split) * PART_STRIDE;
    #pragma unroll
    for (int b = 0; b < 5; ++b)
        #pragma unroll
        for (int r = 0; r < 8; ++r)
            pd[(r + (hi ? 8 : 0)) * XP + b * 16 + l15] = O[b][r];
    if (l15 == 0) {
        #pragma unroll
        for (int r = 0; r < 8; ++r) pd[16 * XP + (r + (hi ? 8 : 0))] = mrow[r];
    }
    if (l15 == 2) {
        #pragma unroll
        for (int r = 0; r < 8; ++r) pd[16 * XP + 16 + (r + (hi ? 8 : 0))] = O[4][r];
    }
}

// ---------------------------------------------------------------------------
// Phase 4: merge NSPLIT partials per 16-row tile, normalize, write f32 out
// ---------------------------------------------------------------------------
__global__ void combine_kernel(const float* __restrict__ part, float* __restrict__ out) {
    int tile = blockIdx.x;
    int e = blockIdx.y * blockDim.x + threadIdx.x;
    if (e >= 16 * DIM) return;
    int row = e / DIM, d = e % DIM;
    const float* pb = part + (size_t)tile * NSPLIT * PART_STRIDE;
    float mg = -__builtin_inff();
    #pragma unroll
    for (int s = 0; s < NSPLIT; ++s)
        mg = fmaxf(mg, pb[s * PART_STRIDE + 16 * XP + row]);
    float osum = 0.f, lsum = 0.f;
    #pragma unroll
    for (int s = 0; s < NSPLIT; ++s) {
        const float* ps = pb + s * PART_STRIDE;
        float w = __expf(ps[16 * XP + row] - mg);
        lsum += w * ps[16 * XP + 16 + row];
        osum += w * ps[row * XP + d];
    }
    out[(size_t)(tile * 16 + row) * DIM + d] = osum / lsum;
}

// ---------------------------------------------------------------------------
extern "C" void kernel_launch(void* const* d_in, const int* in_sizes, int n_in,
                              void* d_out, int out_size, void* d_ws, size_t ws_size,
                              hipStream_t stream) {
    const float* x     = (const float*)d_in[0];
    const float* query = (const float*)d_in[1];
    const float* Wk    = (const float*)d_in[2];
    const float* bk    = (const float*)d_in[3];
    float* out = (float*)d_out;

    char* ws = (char*)d_ws;
    // layout (256B aligned): Kh 12.6MB | qh 1.6MB | xT 10.5MB | partials 21.5MB
    _Float16* Kh    = (_Float16*)(ws);
    _Float16* qh    = (_Float16*)(ws + (size_t)N_PAIRS * DPK * 2);
    _Float16* xT    = (_Float16*)(ws + (size_t)(N_PAIRS + N_QUERY) * DPK * 2);
    float*    partv = (float*) (ws + (size_t)(N_PAIRS + N_QUERY) * DPK * 2
                                   + (size_t)XP * N_PAIRS * 2);

    int prepThreads = (N_PAIRS + N_QUERY) * DPK;     // multiple of 256
    prep_kernel<<<prepThreads / 256, 256, 0, stream>>>(x, query, Wk, bk, Kh, qh);

    xpose_kernel<<<dim3(N_PAIRS / 32, 3), dim3(32, 8), 0, stream>>>(x, xT);

    flash_kernel<<<dim3(N_QUERY / 64, NSPLIT), 128, 0, stream>>>(Kh, qh, xT, partv);

    combine_kernel<<<dim3(N_QUERY / 16, (16 * DIM + 255) / 256), 256, 0, stream>>>(partv, out);
}
